// MHA_962072674722
// MI455X (gfx1250) — matrix operand order repaired and checked by reading, not compiled, not run on hardware
//
#include <hip/hip_runtime.h>
#include <hip/hip_bf16.h>

// ---------------------------------------------------------------------------
// MHA forward for MI455X (gfx1250, wave32, WMMA).
//   Q = X Wq + bq ; K,V = X Wkv + bkv ; flash attention ; out = AO Wo + bo
// All GEMMs + attention matmuls use v_wmma_f32_16x16x32_f16.
// Softmax: row-max via DPP8 + packed f16 max; denominator via WMMA against an
// all-ones tile (every lane holds the row sum -> shuffle-free epilogue).
// ---------------------------------------------------------------------------

typedef _Float16 half16 __attribute__((ext_vector_type(16)));
typedef _Float16 half8  __attribute__((ext_vector_type(8)));
typedef _Float16 half4  __attribute__((ext_vector_type(4)));
typedef _Float16 half2v __attribute__((ext_vector_type(2)));
typedef float    f32x8  __attribute__((ext_vector_type(8)));

union H16 { half16 v; half8 h8[2]; };
union U32H2 { unsigned u; half2v h; };

#define DIM    1024
#define HEADS  16
#define DHEAD  64
#define SEQ    2048
#define BATCH  2
#define MTOT   (BATCH * SEQ)   // 4096 rows

__device__ __forceinline__ f32x8 wmma16(const H16& a, const H16& b, f32x8 c) {
    // D = A(16x32 f16) * B(32x16 f16) + C(16x16 f32)
    return __builtin_amdgcn_wmma_f32_16x16x32_f16(
        /*neg_a=*/false, a.v, /*neg_b=*/false, b.v,
        /*c_mod=*/(short)0, c, /*reuse_a=*/false, /*reuse_b=*/false);
}

// ---------------------------------------------------------------------------
// f32 -> f16 elementwise convert (activations). n4 = count/4.
// ---------------------------------------------------------------------------
__global__ void cvt_f16_kernel(const float* __restrict__ src,
                               _Float16* __restrict__ dst, int n4) {
    int i = blockIdx.x * blockDim.x + threadIdx.x;
    if (i < n4) {
        float4 v = ((const float4*)src)[i];
        half4 o = { (_Float16)v.x, (_Float16)v.y, (_Float16)v.z, (_Float16)v.w };
        ((half4*)dst)[i] = o;
    }
}

// ---------------------------------------------------------------------------
// W[K][N] f32 -> Wt[N][K] f16 (so GEMM B-fragments are K-contiguous per lane).
// 32x32 LDS tile, block (32,8).
// ---------------------------------------------------------------------------
__global__ void transpose_w_kernel(const float* __restrict__ W,
                                   _Float16* __restrict__ Wt, int K, int N) {
    __shared__ _Float16 t[32][33];
    const int tx = threadIdx.x, ty = threadIdx.y;
    const int nb = blockIdx.x * 32, kb = blockIdx.y * 32;
#pragma unroll
    for (int j = 0; j < 4; ++j)
        t[ty + j * 8][tx] = (_Float16)W[(size_t)(kb + ty + j * 8) * N + nb + tx];
    __syncthreads();
#pragma unroll
    for (int j = 0; j < 4; ++j)
        Wt[(size_t)(nb + ty + j * 8) * K + kb + tx] = t[tx][ty + j * 8];
}

// ---------------------------------------------------------------------------
// WMMA GEMM:  C[M][N] = A16[M][K] * Bt16[N][K]^T + bias[N]
// BM=BN=128, BK=64. 256 threads = 8 waves; wave tile 64x32 (4x2 WMMA tiles).
// LDS pitch 72 halves (144B rows -> 16B-aligned, conflict-spread b128 reads).
// ---------------------------------------------------------------------------
template <bool OUTF16>
__global__ __launch_bounds__(256)
void gemm_wmma(const _Float16* __restrict__ A, const _Float16* __restrict__ Bt,
               const float* __restrict__ bias, void* __restrict__ Cv,
               int M, int N, int K, int ldc) {
    constexpr int PIT = 72;
    __shared__ _Float16 As[128 * PIT];
    __shared__ _Float16 Bs[128 * PIT];

    const int tid = threadIdx.x, lane = tid & 31, wave = tid >> 5;
    const int wm = (wave >> 2) * 64;   // 0 / 64
    const int wn = (wave & 3) * 32;    // 0..96
    const int m0 = blockIdx.y * 128, n0 = blockIdx.x * 128;

    f32x8 acc[4][2];
#pragma unroll
    for (int i = 0; i < 4; ++i)
#pragma unroll
        for (int j = 0; j < 2; ++j) acc[i][j] = {};

    const int lrow = tid >> 1;            // 0..127
    const int lseg = (tid & 1) * 32;      // 0 / 32 halves
    const _Float16* ag = A  + (size_t)(m0 + lrow) * K + lseg;
    const _Float16* bg = Bt + (size_t)(n0 + lrow) * K + lseg;
    _Float16* asd = As + lrow * PIT + lseg;
    _Float16* bsd = Bs + lrow * PIT + lseg;

    for (int kb = 0; kb < K; kb += 64) {
        half8 av[4], bv[4];
#pragma unroll
        for (int j = 0; j < 4; ++j) {
            av[j] = *(const half8*)(ag + kb + j * 8);
            bv[j] = *(const half8*)(bg + kb + j * 8);
        }
        __syncthreads();                    // previous tile fully consumed
#pragma unroll
        for (int j = 0; j < 4; ++j) {
            *(half8*)(asd + j * 8) = av[j];
            *(half8*)(bsd + j * 8) = bv[j];
        }
        __syncthreads();
        if (kb + 64 < K) {                  // emits global_prefetch_b8
            __builtin_prefetch(ag + kb + 64, 0, 1);
            __builtin_prefetch(bg + kb + 64, 0, 1);
        }
#pragma unroll
        for (int kk = 0; kk < 2; ++kk) {
            H16 af[4], bf[2];
#pragma unroll
            for (int mt = 0; mt < 4; ++mt) {   // A-frag: row=lane&15, kb=(l>>4)*8
                const _Float16* p =
                    As + (wm + mt * 16 + (lane & 15)) * PIT + kk * 32 + ((lane >> 4) * 8);
                af[mt].h8[0] = *(const half8*)p;
                af[mt].h8[1] = *(const half8*)(p + 16);
            }
#pragma unroll
            for (int nt = 0; nt < 2; ++nt) {   // B-frag: col=lane&15, kb=(l>>4)*8
                const _Float16* p =
                    Bs + (wn + nt * 16 + (lane & 15)) * PIT + kk * 32 + ((lane >> 4) * 8);
                bf[nt].h8[0] = *(const half8*)p;
                bf[nt].h8[1] = *(const half8*)(p + 16);
            }
#pragma unroll
            for (int mt = 0; mt < 4; ++mt)
#pragma unroll
                for (int nt = 0; nt < 2; ++nt)
                    acc[mt][nt] = wmma16(af[mt], bf[nt], acc[mt][nt]);
        }
    }

    // Epilogue: C layout -> lane n = l&15, row m = r + (l>>4)*8 per 16x16 tile.
#pragma unroll
    for (int nt = 0; nt < 2; ++nt) {
        const int col = n0 + wn + nt * 16 + (lane & 15);
        const float bvv = bias[col];
#pragma unroll
        for (int mt = 0; mt < 4; ++mt) {
            const int rowb = m0 + wm + mt * 16 + ((lane >> 4) * 8);
#pragma unroll
            for (int r = 0; r < 8; ++r) {
                float v = acc[mt][nt][r] + bvv;
                if (OUTF16)
                    ((_Float16*)Cv)[(size_t)(rowb + r) * ldc + col] = (_Float16)v;
                else
                    ((float*)Cv)[(size_t)(rowb + r) * ldc + col] = v;
            }
        }
    }
}

// ---------------------------------------------------------------------------
// Flash attention. Block = 256 thr (8 waves) = (b, h, 128 query rows);
// each wave owns 16 query rows. Key tile = 64 keys per iteration.
// Key interleave: S-tile nt, column nn <-> key kt + 4*nn + nt, so each lane
// owns 4 consecutive keys per row -> P spills as conflict-free ds_store_b64.
// V transpose: key-quad repack in registers -> 4x ds_store_b64 per thread.
// Row-max: DPP8 xor-1/2/4 on packed f16 pairs + one grouped bpermute (xor-8).
// Denominator: WMMA vs all-ones tile -> every lane holds the row sum.
// ---------------------------------------------------------------------------
#define DPP8_XOR1 067452301   // lanes [1,0,3,2,5,4,7,6] (octal digits l7..l0)
#define DPP8_XOR2 054761032   // lanes [2,3,0,1,6,7,4,5]
#define DPP8_XOR4 032107654   // lanes [4,5,6,7,0,1,2,3]

__global__ __launch_bounds__(256)
void attn_kernel(const _Float16* __restrict__ Q, const _Float16* __restrict__ KV,
                 _Float16* __restrict__ O) {
    constexpr int PIT = 72;                 // 64 keys + pad, 144B rows
    __shared__ _Float16 Vt[64 * PIT];       // [d][key]
    __shared__ _Float16 Ones[16 * PIT];     // cols 0..63 = 1 in every row
    __shared__ _Float16 Ps[8 * 16 * PIT];   // per-wave 16x64 P

    const int tid = threadIdx.x, lane = tid & 31, wave = tid >> 5;
    const int nqb = SEQ / 128;              // 16
    const int qb = blockIdx.x % nqb;
    const int bh = blockIdx.x / nqb;
    const int h = bh % HEADS, b = bh / HEADS;
    const int q0 = qb * 128 + wave * 16;

    const _Float16* Qb = Q  + (size_t)(b * SEQ) * DIM + h * DHEAD;
    const _Float16* Kb = KV + (size_t)(b * SEQ) * (2 * DIM) + h * DHEAD;
    const _Float16* Vb = Kb + DIM;

    // One-time ones-tile init: B[k][n] = 1 for k<64, all n -> D col n = rowsum
    for (int idx = tid; idx < 16 * PIT; idx += 256) {
        int col = idx % PIT;
        Ones[idx] = (col < 64) ? (_Float16)1.0f : (_Float16)0.0f;
    }

    // Q fragments (2 k-steps over d=64), pre-scaled by 1/sqrt(64)=0.125 (exact).
    H16 qa[2];
    {
        const _Float16* qr = Qb + (size_t)(q0 + (lane & 15)) * DIM;
#pragma unroll
        for (int kk = 0; kk < 2; ++kk) {
            int kbo = kk * 32 + (lane >> 4) * 8;
            H16 t;
            t.h8[0] = *(const half8*)(qr + kbo);
            t.h8[1] = *(const half8*)(qr + kbo + 16);
#pragma unroll
            for (int e = 0; e < 16; ++e)
                qa[kk].v[e] = (_Float16)((float)t.v[e] * 0.125f);
        }
    }

    float mrun[8];
    f32x8 acc[5];                 // acc[0..3]: output d-tiles; acc[4]: denom
#pragma unroll
    for (int r = 0; r < 8; ++r) mrun[r] = -1e30f;
#pragma unroll
    for (int nt = 0; nt < 5; ++nt) acc[nt] = {};

    const int vkq = (tid & 15) * 4;   // key quad base: 0..60
    const int vdg = (tid >> 4) * 4;   // d group of 4:  0..60

    for (int kt = 0; kt < SEQ; kt += 64) {
        __syncthreads();              // all waves done reading previous Vt
        // V-tile transpose: 4x global b64 (rows = keys), register repack,
        // 4x conflict-free ds_store_b64 into Vt[d][key].
        {
            const _Float16* vr = Vb + (size_t)(kt + vkq) * (2 * DIM) + vdg;
            half4 r0 = *(const half4*)(vr);
            half4 r1 = *(const half4*)(vr + 2 * DIM);
            half4 r2 = *(const half4*)(vr + 4 * DIM);
            half4 r3 = *(const half4*)(vr + 6 * DIM);
#pragma unroll
            for (int j = 0; j < 4; ++j) {
                half4 w = { r0[j], r1[j], r2[j], r3[j] };
                *(half4*)(Vt + (vdg + j) * PIT + vkq) = w;
            }
        }
        __syncthreads();

        // S = Qs * K^T : four independent 16x16 tiles; tile nt column nn is
        // global key kt + 4*nn + nt (interleaved mapping).
        f32x8 s[4];
#pragma unroll
        for (int nt = 0; nt < 4; ++nt) s[nt] = {};
#pragma unroll
        for (int kk = 0; kk < 2; ++kk)
#pragma unroll
            for (int nt = 0; nt < 4; ++nt) {
                const _Float16* kr =
                    Kb + (size_t)(kt + 4 * (lane & 15) + nt) * (2 * DIM);
                H16 bf;
                const _Float16* p = kr + kk * 32 + (lane >> 4) * 8;
                bf.h8[0] = *(const half8*)p;
                bf.h8[1] = *(const half8*)(p + 16);
                s[nt] = wmma16(qa[kk], bf, s[nt]);
            }

        // Row-max over 64 keys: local max, then 16-lane reduction on packed
        // f16 pairs: DPP8 xor-1/2/4 (VALU) + one grouped bpermute (xor-8).
        U32H2 pk[4];
#pragma unroll
        for (int i = 0; i < 4; ++i) {
            float e0 = fmaxf(fmaxf(s[0][2 * i],     s[1][2 * i]),
                             fmaxf(s[2][2 * i],     s[3][2 * i]));
            float e1 = fmaxf(fmaxf(s[0][2 * i + 1], s[1][2 * i + 1]),
                             fmaxf(s[2][2 * i + 1], s[3][2 * i + 1]));
            pk[i].h = half2v{ (_Float16)e0, (_Float16)e1 };
        }
#pragma unroll
        for (int i = 0; i < 4; ++i) {
            U32H2 o;
            o.u = __builtin_amdgcn_mov_dpp8(pk[i].u, DPP8_XOR1);
            pk[i].h = __builtin_elementwise_max(pk[i].h, o.h);
            o.u = __builtin_amdgcn_mov_dpp8(pk[i].u, DPP8_XOR2);
            pk[i].h = __builtin_elementwise_max(pk[i].h, o.h);
            o.u = __builtin_amdgcn_mov_dpp8(pk[i].u, DPP8_XOR4);
            pk[i].h = __builtin_elementwise_max(pk[i].h, o.h);
        }
        {   // xor-8 stage: 4 independent bpermutes (one clause, one wait)
            U32H2 o0, o1, o2, o3;
            o0.u = (unsigned)__shfl_xor((int)pk[0].u, 8);
            o1.u = (unsigned)__shfl_xor((int)pk[1].u, 8);
            o2.u = (unsigned)__shfl_xor((int)pk[2].u, 8);
            o3.u = (unsigned)__shfl_xor((int)pk[3].u, 8);
            pk[0].h = __builtin_elementwise_max(pk[0].h, o0.h);
            pk[1].h = __builtin_elementwise_max(pk[1].h, o1.h);
            pk[2].h = __builtin_elementwise_max(pk[2].h, o2.h);
            pk[3].h = __builtin_elementwise_max(pk[3].h, o3.h);
        }

        float al[8];
#pragma unroll
        for (int r = 0; r < 8; ++r) {
            const float tmax = (float)pk[r >> 1].h[r & 1];
            const float mn = fmaxf(mrun[r], tmax);
            al[r] = __expf(mrun[r] - mn);
            mrun[r] = mn;
#pragma unroll
            for (int nt = 0; nt < 4; ++nt)
                s[nt][r] = __expf(s[nt][r] - mn);
        }
#pragma unroll
        for (int nt = 0; nt < 5; ++nt)
#pragma unroll
            for (int r = 0; r < 8; ++r) acc[nt][r] *= al[r];

        // P spill: lane owns keys 4*nn..4*nn+3 per row -> 8x ds_store_b64,
        // conflict-free (lanes 0-15 banks 36r+[0,31], lanes 16-31 +[32,63]).
        _Float16* Pw = Ps + wave * 16 * PIT;
        {
            const int mb = (lane >> 4) * 8, nn = lane & 15;
#pragma unroll
            for (int r = 0; r < 8; ++r) {
                half4 pv = { (_Float16)s[0][r], (_Float16)s[1][r],
                             (_Float16)s[2][r], (_Float16)s[3][r] };
                *(half4*)(Pw + (mb + r) * PIT + nn * 4) = pv;
            }
        }
        asm volatile("s_wait_dscnt 0" ::: "memory");  // wave-local LDS RAW fence

        H16 pf[2];   // A-fragments of P (16 q-rows x 64 keys, 2 k-steps)
#pragma unroll
        for (int kk = 0; kk < 2; ++kk) {
            const _Float16* p =
                Pw + (lane & 15) * PIT + kk * 32 + (lane >> 4) * 8;
            pf[kk].h8[0] = *(const half8*)p;
            pf[kk].h8[1] = *(const half8*)(p + 16);
        }
#pragma unroll
        for (int kk = 0; kk < 2; ++kk) {
            const int kb = kk * 32 + (lane >> 4) * 8;
#pragma unroll
            for (int nt = 0; nt < 4; ++nt) {   // B-frags of V^T: lane = d col
                H16 vf;
                const _Float16* p = Vt + (nt * 16 + (lane & 15)) * PIT + kb;
                vf.h8[0] = *(const half8*)p;
                vf.h8[1] = *(const half8*)(p + 16);
                acc[nt] = wmma16(pf[kk], vf, acc[nt]);
            }
            {   // denominator: acc4 += P @ ones (every column = row sum)
                H16 of;
                const _Float16* p = Ones + (lane & 15) * PIT + kb;
                of.h8[0] = *(const half8*)p;
                of.h8[1] = *(const half8*)(p + 8);
                acc[4] = wmma16(pf[kk], of, acc[4]);
            }
        }
    }

    // Epilogue: every lane holds its rows' denominators in acc[4].
    _Float16* Ob = O + (size_t)(b * SEQ + q0) * DIM + h * DHEAD;
#pragma unroll
    for (int r = 0; r < 8; ++r) {
        const float inv = 1.0f / acc[4][r];
        const int row = (lane >> 4) * 8 + r;
#pragma unroll
        for (int nt = 0; nt < 4; ++nt) {
            const int col = nt * 16 + (lane & 15);
            Ob[(size_t)row * DIM + col] = (_Float16)(acc[nt][r] * inv);
        }
    }
}

// ---------------------------------------------------------------------------
// Launch: convert -> transpose weights -> Q/KV GEMMs -> attention -> out GEMM
// ---------------------------------------------------------------------------
extern "C" void kernel_launch(void* const* d_in, const int* in_sizes, int n_in,
                              void* d_out, int out_size, void* d_ws, size_t ws_size,
                              hipStream_t stream) {
    (void)in_sizes; (void)n_in; (void)out_size; (void)ws_size;
    const float* q   = (const float*)d_in[0];
    const float* Wq  = (const float*)d_in[1];
    const float* bq  = (const float*)d_in[2];
    const float* Wkv = (const float*)d_in[3];
    const float* bkv = (const float*)d_in[4];
    const float* Wo  = (const float*)d_in[5];
    const float* bo  = (const float*)d_in[6];
    float* out = (float*)d_out;

    char* ws = (char*)d_ws;
    _Float16* X16  = (_Float16*)ws; ws += (size_t)MTOT * DIM * 2;        // 8 MB
    _Float16* WqT  = (_Float16*)ws; ws += (size_t)DIM * DIM * 2;         // 2 MB
    _Float16* WkvT = (_Float16*)ws; ws += (size_t)(2 * DIM) * DIM * 2;   // 4 MB
    _Float16* WoT  = (_Float16*)ws; ws += (size_t)DIM * DIM * 2;         // 2 MB
    _Float16* Q16  = (_Float16*)ws; ws += (size_t)MTOT * DIM * 2;        // 8 MB
    _Float16* KV16 = (_Float16*)ws; ws += (size_t)MTOT * 2 * DIM * 2;    // 16 MB
    _Float16* AO16 = (_Float16*)ws; ws += (size_t)MTOT * DIM * 2;        // 8 MB

    cvt_f16_kernel<<<(MTOT * DIM / 4 + 255) / 256, 256, 0, stream>>>(
        q, X16, MTOT * DIM / 4);
    transpose_w_kernel<<<dim3(DIM / 32, DIM / 32), dim3(32, 8), 0, stream>>>(
        Wq, WqT, DIM, DIM);
    transpose_w_kernel<<<dim3(2 * DIM / 32, DIM / 32), dim3(32, 8), 0, stream>>>(
        Wkv, WkvT, DIM, 2 * DIM);
    transpose_w_kernel<<<dim3(DIM / 32, DIM / 32), dim3(32, 8), 0, stream>>>(
        Wo, WoT, DIM, DIM);

    gemm_wmma<true><<<dim3(DIM / 128, MTOT / 128), 256, 0, stream>>>(
        X16, WqT, bq, (void*)Q16, MTOT, DIM, DIM, DIM);
    gemm_wmma<true><<<dim3(2 * DIM / 128, MTOT / 128), 256, 0, stream>>>(
        X16, WkvT, bkv, (void*)KV16, MTOT, 2 * DIM, DIM, 2 * DIM);

    attn_kernel<<<BATCH * HEADS * (SEQ / 128), 256, 0, stream>>>(Q16, KV16, AO16);

    gemm_wmma<false><<<dim3(DIM / 128, MTOT / 128), 256, 0, stream>>>(
        AO16, WoT, bo, (void*)out, MTOT, DIM, DIM, DIM);
}
